// Fused_86998857548375
// MI455X (gfx1250) — matrix operand order repaired and checked by reading, not compiled, not run on hardware
//
#include <hip/hip_runtime.h>
#include <stdint.h>

#define NV 50000
#define NE 25000
#define NM 300000
#define DIM 256
#define QDIM 64
#define NCLS 10

#if __has_include(<hip/amd_detail/amd_gfx1250_TDM.h>)
#define TDM_6ARG 1
#else
#define TDM_6ARG 0
#endif

typedef __attribute__((ext_vector_type(16))) __bf16 v16bf;
typedef __attribute__((ext_vector_type(8)))  float  v8f;
typedef __attribute__((ext_vector_type(4)))  unsigned int v4u;
typedef __attribute__((ext_vector_type(8)))  int v8i;
typedef __attribute__((ext_vector_type(4)))  int v4i;
typedef __attribute__((address_space(3))) const void* lds_cptr;

union FragB16 {
    v16bf v;
    uint4 q[2];
    unsigned short u[16];
};

__device__ __forceinline__ unsigned short f2bf(float x) {
    unsigned int u = __float_as_uint(x);
    unsigned int r = u + 0x7FFFu + ((u >> 16) & 1u);   // RNE
    return (unsigned short)(r >> 16);
}

__device__ __forceinline__ v16bf load_frag(const unsigned short* p) {
    FragB16 f;
    f.q[0] = ((const uint4*)p)[0];
    f.q[1] = ((const uint4*)p)[1];
    return f.v;
}

__device__ __forceinline__ unsigned lds_offset_of(const void* p) {
    return (unsigned)(unsigned long long)(lds_cptr)p;
}

__device__ __forceinline__ float atomicMaxFloat(float* addr, float val) {
    if (val >= 0.0f)
        return __int_as_float(atomicMax((int*)addr, __float_as_int(val)));
    else
        return __uint_as_float(atomicMin((unsigned int*)addr, __float_as_uint(val)));
}

// ---- TDM: async 2D bf16 tile (32 cols x tile1 rows) global -> LDS ----------
// Padded LDS layout: every 16 dwords (64B row) followed by 4 pad dwords
// -> 80B row stride, keeps 16B alignment for fragment b128 reads.
__device__ __forceinline__ void tdm_load_2d(unsigned lds_addr, const void* gaddr,
                                            int dim0, int dim1, int tile1) {
    unsigned long long ga = (unsigned long long)gaddr;
    v4u g0;
    g0[0] = 1u;                                   // count=1, user descriptor
    g0[1] = lds_addr;                             // LDS byte address
    g0[2] = (unsigned)ga;                         // global_addr[31:0]
    g0[3] = (unsigned)((ga >> 32) & 0x1FFFFFFu) | (2u << 30);  // addr[56:32] | type=2
    v8i g1;
    g1[0] = (int)((1u << 16) |                    // data_size = 2 bytes (bf16)
                  (1u << 20) |                    // pad_enable
                  (3u << 22) |                    // pad_interval: 16 dwords
                  (3u << 25));                    // pad_amount: 4 dwords
    g1[1] = (int)(((unsigned)dim0 & 0xFFFFu) << 16);              // tensor_dim0 lo
    g1[2] = (int)(((unsigned)dim0 >> 16) |
                  (((unsigned)dim1 & 0xFFFFu) << 16));            // dim0 hi | dim1 lo
    g1[3] = (int)((((unsigned)dim1 >> 16) & 0xFFFFu) |
                  (32u << 16));                                   // dim1 hi | tile_dim0=32
    g1[4] = tile1;                                // tile_dim1 | tile_dim2=0
    g1[5] = dim0;                                 // tensor_dim0_stride (elements)
    g1[6] = 0;
    g1[7] = 0;
    v4i g2 = {0, 0, 0, 0};
    v4i g3 = {0, 0, 0, 0};
#if TDM_6ARG
    v8i g4 = {0, 0, 0, 0, 0, 0, 0, 0};
    __builtin_amdgcn_tensor_load_to_lds(g0, g1, g2, g3, g4, 0);
#else
    __builtin_amdgcn_tensor_load_to_lds(g0, g1, g2, g3, 0);
#endif
}

// ---------------- fill ----------------
__global__ void fill_kernel(float* __restrict__ p, float v, long long n) {
    long long i = (long long)blockIdx.x * blockDim.x + threadIdx.x;
    long long stride = (long long)gridDim.x * blockDim.x;
    for (; i < n; i += stride) p[i] = v;
}

__global__ void f32_to_bf16_kernel(const float* __restrict__ in,
                                   unsigned short* __restrict__ out, long long n) {
    long long i = (long long)blockIdx.x * blockDim.x + threadIdx.x;
    long long stride = (long long)gridDim.x * blockDim.x;
    for (; i < n; i += stride) out[i] = f2bf(in[i]);
}

// ---------------- weight packing into bf16 B-fragments ----------------
// B-fragment (32x16 KxN, 16-bit): lane l -> col n=l&15, half h=l>>4,
// element j -> k = 8h+j (j<8), 16+8h+(j-8) (j>=8). B[k][n] = W[n][k]. 16 contiguous ushorts/lane.
__global__ void pack_w_kernel(const float* __restrict__ W, unsigned short* __restrict__ Wp,
                              int dout, int din) {
    int ksteps  = din >> 5;
    int ctTiles = (dout + 15) >> 4;
    int total   = ctTiles * ksteps * 32;
    int t = blockIdx.x * blockDim.x + threadIdx.x;
    if (t >= total) return;
    int lane = t & 31;
    int tile = t >> 5;
    int ks = tile % ksteps;
    int ct = tile / ksteps;
    int n = ct * 16 + (lane & 15);
    int h = lane >> 4;
    unsigned short* out = Wp + (size_t)t * 16;
#pragma unroll
    for (int j = 0; j < 16; ++j) {
        int k = ks * 32 + (j < 8 ? h * 8 + j : 16 + h * 8 + (j - 8));
        float v = (n < dout) ? W[(size_t)n * din + k] : 0.0f;
        out[j] = f2bf(v);
    }
}

// ---------------- TDM double-buffered bf16 WMMA GEMM: C = A*W^T + bias ------
// A: [Mrows,K] bf16 row-major. 256 threads = 8 waves. Block tile:
// (RT*16) rows x (CT*NACC*16) cols, CT = 8/RT. Wave wid: row-tile rt=wid/CT,
// col-wave cw=wid%CT, accumulates NACC tiles ct = cw + CT*a. All WMMAs
// unconditional (EXEC all-ones). A staged by tensor_load_to_lds (TENSORcnt).
template <int RT, int NACC>
__global__ __launch_bounds__(256)
void gemm_tdm_kernel(const unsigned short* __restrict__ Abf,
                     const unsigned short* __restrict__ Wp,
                     const float* __restrict__ bias, float* __restrict__ C,
                     int Mrows, int K, int dout, int Cld) {
    constexpr int CT   = 8 / RT;
    constexpr int ROWB = 80;               // 64B data + 16B TDM pad per row
    constexpr int BUFB = RT * 16 * ROWB;   // bytes per LDS buffer
    __shared__ __align__(16) unsigned char lds[2 * BUFB];

    const int ksteps = K >> 5;
    const int tid  = threadIdx.x;
    const int wid  = tid >> 5;
    const int lane = tid & 31;
    const int rt   = wid / CT;
    const int cw   = wid % CT;
    const int row0 = blockIdx.x * (RT * 16);

    const unsigned lds_base = lds_offset_of(lds);
    const unsigned short* atile = Abf + (size_t)row0 * K;
    const int dim1 = Mrows - row0;         // rows OOB in TDM read as zero

    if (wid == 0) tdm_load_2d(lds_base, atile, K, dim1, RT * 16);

    v8f acc[NACC];
#pragma unroll
    for (int a = 0; a < NACC; ++a) acc[a] = (v8f){};

    const int r = lane & 15;
    const int h = lane >> 4;
    const unsigned char* fr = lds + (rt * 16 + r) * ROWB + h * 16;

    for (int ks = 0; ks < ksteps; ++ks) {
        __syncthreads();                   // everyone done reading buf[(ks+1)&1]
        if (wid == 0) {
            if (ks + 1 < ksteps) {
                tdm_load_2d(lds_base + (unsigned)(((ks + 1) & 1) * BUFB),
                            atile + (ks + 1) * 32, K, dim1, RT * 16);
                __builtin_amdgcn_s_wait_tensorcnt(1);   // buf[ks&1] complete
            } else {
                __builtin_amdgcn_s_wait_tensorcnt(0);
            }
        }
        __syncthreads();                   // buf[ks&1] visible to all waves
        const unsigned char* p = fr + (ks & 1) * BUFB;
        FragB16 af;
        af.q[0] = *(const uint4*)p;          // k = 8h .. 8h+7
        af.q[1] = *(const uint4*)(p + 32);   // k = 16+8h .. 16+8h+7
#pragma unroll
        for (int a = 0; a < NACC; ++a) {
            int ct = cw + CT * a;
            v16bf b = load_frag(Wp + ((size_t)(ct * ksteps + ks) * 32 + lane) * 16);
            acc[a] = __builtin_amdgcn_wmma_f32_16x16x32_bf16(false, af.v, false, b,
                                                             (short)0, acc[a], false, false);
        }
    }

    // C/D layout: VGPR g, lane l -> row = g + 8*(l>>4), col = l&15 within tile
    const int cn = lane & 15;
#pragma unroll
    for (int a = 0; a < NACC; ++a) {
        int ct  = cw + CT * a;
        int col = ct * 16 + cn;
        if (col >= dout) continue;
        float bv = bias[col];
#pragma unroll
        for (int g = 0; g < 8; ++g) {
            int row = row0 + rt * 16 + g + h * 8;
            if (row < Mrows) C[(size_t)row * Cld + col] = acc[a][g] + bv;
        }
    }
}

// ---------------- vertex -> hyperedge weighted scatter ----------------
__global__ void scatter_edge_kernel(const float* __restrict__ Wh,
                                    const float* __restrict__ v_reg_weight,
                                    const float* __restrict__ e_reg_sum,
                                    const int* __restrict__ in_src,
                                    const int* __restrict__ in_dst,
                                    float* __restrict__ feat_e) {
    int m = blockIdx.x;
    int s = in_src[m];
    int d = in_dst[m];
    float w = v_reg_weight[s] / e_reg_sum[d];
    int c = threadIdx.x;
    atomicAdd(&feat_e[(size_t)d * DIM + c], w * Wh[(size_t)s * DIM + c]);
}

// ---------------- attention logits: one wave per incidence ----------------
__global__ void attn_kernel(const float* __restrict__ ke, const float* __restrict__ qv,
                            const int* __restrict__ in_src, const int* __restrict__ in_dst,
                            float* __restrict__ attn, int Mtot) {
    int wid  = threadIdx.x >> 5;
    int lane = threadIdx.x & 31;
    int m = blockIdx.x * 8 + wid;
    if (m >= Mtot) return;
    int s = in_src[m];
    int d = in_dst[m];
    const float* kp = ke + (size_t)d * QDIM;
    const float* qp = qv + (size_t)s * QDIM;
    float acc = kp[lane] * qp[lane] + kp[lane + 32] * qp[lane + 32];
#pragma unroll
    for (int off = 16; off > 0; off >>= 1)
        acc += __shfl_xor(acc, off, 32);
    if (lane == 0) {
        float a = (acc > 0.0f) ? acc : 0.01f * acc;   // leaky_relu
        attn[m] = a * 0.125f;                          // 1/sqrt(64)
    }
}

__global__ void segmax_kernel(const float* __restrict__ attn, const int* __restrict__ in_src,
                              float* __restrict__ mbuf, int Mtot) {
    int m = blockIdx.x * blockDim.x + threadIdx.x;
    if (m < Mtot) atomicMaxFloat(&mbuf[in_src[m]], attn[m]);
}

__global__ void fixup_kernel(float* __restrict__ mbuf, int n) {
    int i = blockIdx.x * blockDim.x + threadIdx.x;
    if (i < n) {
        float v = mbuf[i];
        if (__float_as_uint(v) == 0xFF800000u) mbuf[i] = 0.0f;   // -inf -> 0
    }
}

__global__ void pdenom_kernel(const float* __restrict__ attn, const int* __restrict__ in_src,
                              const float* __restrict__ mbuf, float* __restrict__ p,
                              float* __restrict__ denom, int Mtot) {
    int m = blockIdx.x * blockDim.x + threadIdx.x;
    if (m < Mtot) {
        int s = in_src[m];
        float pv = expf(attn[m] - mbuf[s]);
        p[m] = pv;
        atomicAdd(&denom[s], pv);
    }
}

// ---------------- hyperedge -> vertex weighted aggregation ----------------
__global__ void agg_kernel(const float* __restrict__ ve, const float* __restrict__ p,
                           const int* __restrict__ in_src, const int* __restrict__ in_dst,
                           float* __restrict__ agg) {
    int m = blockIdx.x;
    int s = in_src[m];
    int d = in_dst[m];
    float pv = p[m];
    int c = threadIdx.x;
    atomicAdd(&agg[(size_t)s * DIM + c], pv * ve[(size_t)d * DIM + c]);
}

__global__ void finalize_kernel(float* __restrict__ fv, const float* __restrict__ denom,
                                long long n) {
    long long i = (long long)blockIdx.x * blockDim.x + threadIdx.x;
    if (i < n) {
        int r = (int)(i >> 8);
        fv[i] = fv[i] / fmaxf(denom[r], 1e-20f);
    }
}

// =====================================================================

static inline size_t alignup256(size_t x) { return (x + 255) & ~(size_t)255; }
static inline size_t packed_bytes(int dout, int din) {
    return (size_t)((dout + 15) / 16) * (din / 32) * 512 * sizeof(unsigned short);
}

extern "C" void kernel_launch(void* const* d_in, const int* in_sizes, int n_in,
                              void* d_out, int out_size, void* d_ws, size_t ws_size,
                              hipStream_t stream) {
    const float* vfeat        = (const float*)d_in[0];
    const float* v_reg_weight = (const float*)d_in[2];
    const float* e_reg_sum    = (const float*)d_in[5];
    const float* W_vtx1 = (const float*)d_in[6];  const float* b_vtx1 = (const float*)d_in[7];
    const float* W_ve   = (const float*)d_in[8];  const float* b_ve   = (const float*)d_in[9];
    const float* W_qv   = (const float*)d_in[10]; const float* b_qv   = (const float*)d_in[11];
    const float* W_ke   = (const float*)d_in[12]; const float* b_ke   = (const float*)d_in[13];
    const float* W_vale = (const float*)d_in[14]; const float* b_vale = (const float*)d_in[15];
    const float* W_cls  = (const float*)d_in[16]; const float* b_cls  = (const float*)d_in[17];
    const int* in_src = (const int*)d_in[18];
    const int* in_dst = (const int*)d_in[19];

    float* out_featv = (float*)d_out;                       // [NV, 256]
    float* out_feate = out_featv + (size_t)NV * DIM;        // [NE, 256]
    float* out_pred  = out_feate + (size_t)NE * DIM;        // [NV, 10]

    // ---- workspace bump allocator ----
    char* ws = (char*)d_ws;
    size_t off = 0;
    auto grab = [&](size_t bytes) -> char* {
        char* p = ws + off;
        off += alignup256(bytes);
        return p;
    };
    float* featv = (float*)grab((size_t)NV * DIM * 4);
    float* Wh    = (float*)grab((size_t)NV * DIM * 4);
    float* ke    = (float*)grab((size_t)NE * QDIM * 4);
    float* ve    = (float*)grab((size_t)NE * DIM * 4);
    float* qv    = (float*)grab((size_t)NV * QDIM * 4);
    float* attn  = (float*)grab((size_t)NM * 4);
    float* pbuf  = (float*)grab((size_t)NM * 4);
    float* mbuf  = (float*)grab((size_t)NV * 4);
    float* denom = (float*)grab((size_t)NV * 4);
    unsigned short* vfeat_bf = (unsigned short*)grab((size_t)NV * DIM * 2);
    unsigned short* featv_bf = (unsigned short*)grab((size_t)NV * DIM * 2);
    unsigned short* feate_bf = (unsigned short*)grab((size_t)NE * DIM * 2);
    unsigned short* fvout_bf = (unsigned short*)grab((size_t)NV * DIM * 2);
    unsigned short* Wp_vtx1 = (unsigned short*)grab(packed_bytes(DIM, DIM));
    unsigned short* Wp_ve   = (unsigned short*)grab(packed_bytes(DIM, DIM));
    unsigned short* Wp_qv   = (unsigned short*)grab(packed_bytes(QDIM, DIM));
    unsigned short* Wp_ke   = (unsigned short*)grab(packed_bytes(QDIM, DIM));
    unsigned short* Wp_vale = (unsigned short*)grab(packed_bytes(DIM, DIM));
    unsigned short* Wp_cls  = (unsigned short*)grab(packed_bytes(NCLS, DIM));

    const float NEG_INF = -__builtin_inff();

    // ---- init accumulators ----
    fill_kernel<<<2048, 256, 0, stream>>>(out_featv, 0.0f, (long long)NV * DIM);
    fill_kernel<<<2048, 256, 0, stream>>>(out_feate, 0.0f, (long long)NE * DIM);
    fill_kernel<<<256, 256, 0, stream>>>(denom, 0.0f, (long long)NV);
    fill_kernel<<<256, 256, 0, stream>>>(mbuf, NEG_INF, (long long)NV);

    // ---- pack weights to bf16 fragments ----
    auto pack = [&](const float* W, unsigned short* Wp, int dout, int din) {
        int total = ((dout + 15) / 16) * (din / 32) * 32;
        pack_w_kernel<<<(total + 255) / 256, 256, 0, stream>>>(W, Wp, dout, din);
    };
    pack(W_vtx1, Wp_vtx1, DIM, DIM);
    pack(W_ve,   Wp_ve,   DIM, DIM);
    pack(W_qv,   Wp_qv,   QDIM, DIM);
    pack(W_ke,   Wp_ke,   QDIM, DIM);
    pack(W_vale, Wp_vale, DIM, DIM);
    pack(W_cls,  Wp_cls,  NCLS, DIM);

    auto cvt = [&](const float* in, unsigned short* out, long long n) {
        f32_to_bf16_kernel<<<2048, 256, 0, stream>>>(in, out, n);
    };

    // 1) feat_v = vfeat @ W_vtx1^T + b
    cvt(vfeat, vfeat_bf, (long long)NV * DIM);
    gemm_tdm_kernel<2, 4><<<(NV + 31) / 32, 256, 0, stream>>>(
        vfeat_bf, Wp_vtx1, b_vtx1, featv, NV, DIM, DIM, DIM);
    // 2) Wh = feat_v @ W_ve^T + b
    cvt(featv, featv_bf, (long long)NV * DIM);
    gemm_tdm_kernel<2, 4><<<(NV + 31) / 32, 256, 0, stream>>>(
        featv_bf, Wp_ve, b_ve, Wh, NV, DIM, DIM, DIM);
    // 3) feat_e = segment_sum(w * Wh[src] -> dst)
    scatter_edge_kernel<<<NM, 256, 0, stream>>>(Wh, v_reg_weight, e_reg_sum,
                                                in_src, in_dst, out_feate);
    // 4) k = feat_e @ W_ke^T ; v = feat_e @ W_vale^T ; q = feat_v @ W_qv^T
    cvt(out_feate, feate_bf, (long long)NE * DIM);
    gemm_tdm_kernel<2, 1><<<(NE + 31) / 32, 256, 0, stream>>>(
        feate_bf, Wp_ke, b_ke, ke, NE, DIM, QDIM, QDIM);
    gemm_tdm_kernel<2, 4><<<(NE + 31) / 32, 256, 0, stream>>>(
        feate_bf, Wp_vale, b_vale, ve, NE, DIM, DIM, DIM);
    gemm_tdm_kernel<2, 1><<<(NV + 31) / 32, 256, 0, stream>>>(
        featv_bf, Wp_qv, b_qv, qv, NV, DIM, QDIM, QDIM);
    // 5) attention logits, segment softmax over in_src
    attn_kernel<<<(NM + 7) / 8, 256, 0, stream>>>(ke, qv, in_src, in_dst, attn, NM);
    segmax_kernel<<<(NM + 255) / 256, 256, 0, stream>>>(attn, in_src, mbuf, NM);
    fixup_kernel<<<(NV + 255) / 256, 256, 0, stream>>>(mbuf, NV);
    pdenom_kernel<<<(NM + 255) / 256, 256, 0, stream>>>(attn, in_src, mbuf, pbuf, denom, NM);
    // 6) agg into out_featv, then normalize in place
    agg_kernel<<<NM, 256, 0, stream>>>(ve, pbuf, in_src, in_dst, out_featv);
    finalize_kernel<<<(int)(((long long)NV * DIM + 255) / 256), 256, 0, stream>>>(
        out_featv, denom, (long long)NV * DIM);
    // 7) pred = feat_v_out @ W_cls^T + b_cls
    cvt(out_featv, fvout_bf, (long long)NV * DIM);
    gemm_tdm_kernel<8, 1><<<(NV + 127) / 128, 256, 0, stream>>>(
        fvout_bf, Wp_cls, b_cls, out_pred, NV, DIM, NCLS, NCLS);
}
